// GCN_40226663694886
// MI455X (gfx1250) — matrix operand order repaired and checked by reading, not compiled, not run on hardware
//
#include <hip/hip_runtime.h>
#include <hip/hip_bf16.h>

typedef _Float16 half_t;
typedef __attribute__((ext_vector_type(16))) _Float16 v16h;
typedef __attribute__((ext_vector_type(8)))  _Float16 v8h;
typedef __attribute__((ext_vector_type(8)))  float    v8f;

// ---------------- problem constants ----------------
#define NN   50000      // nodes
#define NE   150000     // edges
#define NE2  200000     // edges + self loops
#define KIN  5
#define CH   64
#define XD   325        // concat feature dim
#define KPAD 352        // 325 padded to multiple of 32
#define HEADS 2
#define NCOL 1300       // 2*HEADS*OUT = Wl|Wr fused output columns
#define MTILES 3125     // 50000/16

// ---------------- workspace layout (float elements) ----------------
#define OFF_ACC_SUM   0ull
#define OFF_ACC_MEAN  3200000ull
#define OFF_ACC_NE    6400000ull
#define OFF_CNT       9600000ull
#define OFF_EWSUM     9650000ull
#define OFF_ALPHASUM  9650016ull
#define OFF_OPRE      9750016ull
// zero block = [0, 26000016)
#define OFF_ACC_MAX   26000016ull
#define OFF_ACC_NEM   29200016ull
#define OFF_ALPHAMAX  32400016ull
// -inf block = [26000016, 32500016)
#define OFF_ALPHAE    32500016ull
#define OFF_XLR       32900016ull
#define OFF_HALF      97900016ull   // f16 region starts here (16B aligned)
// f16 elements from OFF_HALF: x_pad [NN*KPAD], then Wt [NCOL*KPAD]

// ---------------- helpers ----------------
__device__ __forceinline__ void atomicMaxF(float* addr, float val) {
  if (val >= 0.0f) atomicMax((int*)addr, __float_as_int(val));
  else             atomicMin((unsigned int*)addr, __float_as_uint(val));
}

__device__ __forceinline__ void edge2_sd(const int* __restrict__ ei, int e2, int& s, int& d) {
  if (e2 < NE) { s = ei[e2]; d = ei[NE + e2]; }
  else         { s = e2 - NE; d = e2 - NE; }
}

// ---------------- kernels ----------------
__global__ void fill_kernel(float* __restrict__ p, unsigned long long n, float v) {
  unsigned long long i = (unsigned long long)blockIdx.x * blockDim.x + threadIdx.x;
  unsigned long long st = (unsigned long long)gridDim.x * blockDim.x;
  for (; i < n; i += st) p[i] = v;
}

__global__ void ew_sum_kernel(const float* __restrict__ ew, float* __restrict__ out) {
  __shared__ float s[256];
  int t = blockIdx.x * 256 + threadIdx.x;
  s[threadIdx.x] = (t < NE) ? ew[t] : 0.0f;
  __syncthreads();
  for (int off = 128; off > 0; off >>= 1) {
    if (threadIdx.x < off) s[threadIdx.x] += s[threadIdx.x + off];
    __syncthreads();
  }
  if (threadIdx.x == 0) atomicAdd(out, s[0]);
}

// per (edge, channel): compute 5 GCN messages and scatter with atomics
__global__ void gcn_edge_kernel(const float* __restrict__ h, const int* __restrict__ ei,
                                const float* __restrict__ ew,
                                const float* __restrict__ Wsum, const float* __restrict__ Wmean,
                                const float* __restrict__ Wmax, const float* __restrict__ Wne,
                                const float* __restrict__ Wnem,
                                float* __restrict__ acc_sum, float* __restrict__ acc_mean,
                                float* __restrict__ acc_max, float* __restrict__ acc_ne,
                                float* __restrict__ acc_nem, float* __restrict__ cnt) {
  long long t = (long long)blockIdx.x * blockDim.x + threadIdx.x;
  if (t >= (long long)NE * CH) return;
  int e = (int)(t / CH), c = (int)(t % CH);
  int s = ei[e], d = ei[NE + e];
  float w = ew[e];
  float hs0 = h[s * 5 + 0], hs1 = h[s * 5 + 1], hs2 = h[s * 5 + 2],
        hs3 = h[s * 5 + 3], hs4 = h[s * 5 + 4];
  #define DOT(W) (hs0*W[c] + hs1*W[64+c] + hs2*W[128+c] + hs3*W[192+c] + hs4*W[256+c])
  float msum  = DOT(Wsum);
  float mmean = DOT(Wmean);
  float mmax  = DOT(Wmax);
  float mne   = DOT(Wne);
  float mnem  = DOT(Wnem);
  #undef DOT
  size_t o = (size_t)d * CH + c;
  atomicAdd(&acc_sum[o],  msum * w);
  atomicAdd(&acc_mean[o], mmean * w);
  atomicMaxF(&acc_max[o], mmax * w);
  atomicAdd(&acc_ne[o],   mne);
  atomicMaxF(&acc_nem[o], mnem);
  if (c == 0) atomicAdd(&cnt[d], 1.0f);
}

// per (node, padded column): assemble x row, add biases, convert to f16 (zero pad K 325..351)
__global__ void node_x_kernel(const float* __restrict__ edge_num,
                              const float* __restrict__ acc_sum, const float* __restrict__ acc_mean,
                              const float* __restrict__ acc_max, const float* __restrict__ acc_ne,
                              const float* __restrict__ acc_nem, const float* __restrict__ cnt,
                              const float* __restrict__ bsum, const float* __restrict__ bmean,
                              const float* __restrict__ bmax, const float* __restrict__ bne,
                              const float* __restrict__ bnem,
                              half_t* __restrict__ xpad) {
  long long t = (long long)blockIdx.x * blockDim.x + threadIdx.x;
  if (t >= (long long)NN * KPAD) return;
  int n = (int)(t / KPAD), col = (int)(t % KPAD);
  float v = 0.0f;
  if (col < 64) {
    v = acc_sum[(size_t)n * CH + col] + bsum[col];
  } else if (col < 128) {
    int c = col - 64;
    float cc = cnt[n]; cc = cc < 1.0f ? 1.0f : cc;
    v = acc_mean[(size_t)n * CH + c] / cc + bmean[c];
  } else if (col < 192) {
    int c = col - 128;
    float m = acc_max[(size_t)n * CH + c];
    if (!(m > -3.0e38f)) m = 0.0f;   // empty segment -> 0
    v = m + bmax[c];
  } else if (col < 256) {
    int c = col - 192;
    v = acc_ne[(size_t)n * CH + c] + bne[c];
  } else if (col < 320) {
    int c = col - 256;
    float m = acc_nem[(size_t)n * CH + c];
    if (!(m > -3.0e38f)) m = 0.0f;
    v = m + bnem[c];
  } else if (col < XD) {
    v = edge_num[n * 5 + (col - 320)];
  }
  xpad[t] = (half_t)v;
}

// transpose+convert Wl|Wr into Wt[NCOL][KPAD] f16, zero-padded in K
__global__ void wt_kernel(const float* __restrict__ Wl, const float* __restrict__ Wr,
                          half_t* __restrict__ Wt) {
  long long t = (long long)blockIdx.x * blockDim.x + threadIdx.x;
  if (t >= (long long)NCOL * KPAD) return;
  int col = (int)(t / KPAD), k = (int)(t % KPAD);
  float v = 0.0f;
  if (k < XD) v = (col < 650) ? Wl[(size_t)k * 650 + col] : Wr[(size_t)k * 650 + (col - 650)];
  Wt[t] = (half_t)v;
}

// fused GEMM: xlr[N,1300] = x_pad[N,352] @ (Wl|Wr) via v_wmma_f32_16x16x32_f16
// one wave computes a 16x64 output strip (4 accumulators, A reused)
__global__ __launch_bounds__(256) void gemm_wmma_kernel(const half_t* __restrict__ X,
                                                        const half_t* __restrict__ Wt,
                                                        float* __restrict__ Y) {
  const int lane = threadIdx.x & 31;
  const int wave = threadIdx.x >> 5;
  const int m_tile = blockIdx.x * 8 + wave;
  if (m_tile >= MTILES) return;            // uniform per wave: EXEC stays all-ones for WMMA
  const int n_base = blockIdx.y * 64;
  const int r = lane & 15;
  const int hid = lane >> 4;
  const int row = m_tile * 16 + r;
  const half_t* __restrict__ xrow = X + (size_t)row * KPAD;

  v8f acc0 = {}, acc1 = {}, acc2 = {}, acc3 = {};

  for (int kk = 0; kk < KPAD; kk += 32) {
    __builtin_prefetch(xrow + kk + 64, 0, 1);   // global_prefetch_b8
    // A tile: 16-bit A 16x32 layout -> elems 0..7 = K[kk+hid*8..], elems 8..15 = K[kk+16+hid*8..]
    v16h a;
    {
      const v8h a0 = *(const v8h*)(xrow + kk + hid * 8);
      const v8h a1 = *(const v8h*)(xrow + kk + 16 + hid * 8);
      #pragma unroll
      for (int i = 0; i < 8; ++i) { a[i] = a0[i]; a[i + 8] = a1[i]; }
    }
    #pragma unroll
    for (int nt = 0; nt < 4; ++nt) {
      int col = n_base + nt * 16 + r;
      if (col > NCOL - 1) col = NCOL - 1;   // clamp load; store is guarded
      // B tile: 16-bit B 32x16 -> elems 0..7 = K[kk+hid*8..], elems 8..15 = K[kk+16+hid*8..]
      const half_t* __restrict__ wcol = Wt + (size_t)col * KPAD + kk + hid * 8;
      v16h b;
      const v8h b0 = *(const v8h*)(wcol);
      const v8h b1 = *(const v8h*)(wcol + 16);
      #pragma unroll
      for (int i = 0; i < 8; ++i) { b[i] = b0[i]; b[i + 8] = b1[i]; }
      v8f c = (nt == 0) ? acc0 : (nt == 1) ? acc1 : (nt == 2) ? acc2 : acc3;
      c = __builtin_amdgcn_wmma_f32_16x16x32_f16(false, a, false, b, (short)0, c, false, false);
      if (nt == 0) acc0 = c; else if (nt == 1) acc1 = c; else if (nt == 2) acc2 = c; else acc3 = c;
    }
  }
  // D layout: VGPR j -> row = m_tile*16 + j + hid*8, col = n_base + nt*16 + r
  #pragma unroll
  for (int nt = 0; nt < 4; ++nt) {
    const int col = n_base + nt * 16 + r;
    if (col < NCOL) {
      const v8f c = (nt == 0) ? acc0 : (nt == 1) ? acc1 : (nt == 2) ? acc2 : acc3;
      #pragma unroll
      for (int j = 0; j < 8; ++j) {
        Y[(size_t)(m_tile * 16 + j + hid * 8) * NCOL + col] = c[j];
      }
    }
  }
}

// per (edge2, head): GATv2 attention logit + running segment max
__global__ void alpha1_kernel(const int* __restrict__ ei, const float* __restrict__ ew,
                              const float* __restrict__ ewsum,
                              const float* __restrict__ xlr, const float* __restrict__ We,
                              const float* __restrict__ att,
                              float* __restrict__ alpha_e, float* __restrict__ alpha_max) {
  int t = blockIdx.x * blockDim.x + threadIdx.x;
  if (t >= NE2 * HEADS) return;
  int e2 = t >> 1, hh = t & 1;
  int s, d; edge2_sd(ei, e2, s, d);
  float ea = (e2 < NE) ? ew[e2] : (ewsum[0] * (1.0f / (float)NE));
  const float* __restrict__ xls = xlr + (size_t)s * NCOL + hh * XD;
  const float* __restrict__ xrd = xlr + (size_t)d * NCOL + 650 + hh * XD;
  const float* __restrict__ wep = We + hh * XD;
  const float* __restrict__ atp = att + hh * XD;
  float acc = 0.0f;
  for (int c = 0; c < XD; ++c) {
    float v = xls[c] + xrd[c] + ea * wep[c];
    v = (v > 0.0f) ? v : 0.2f * v;          // leaky_relu(.,0.2)
    acc += atp[c] * v;
  }
  alpha_e[t] = acc;
  atomicMaxF(&alpha_max[d * HEADS + hh], acc);
}

// per (edge2, head): exp(alpha - max), accumulate softmax denominator
__global__ void alpha2_kernel(const int* __restrict__ ei,
                              float* __restrict__ alpha_e, const float* __restrict__ alpha_max,
                              float* __restrict__ alpha_sum) {
  int t = blockIdx.x * blockDim.x + threadIdx.x;
  if (t >= NE2 * HEADS) return;
  int e2 = t >> 1, hh = t & 1;
  int s, d; edge2_sd(ei, e2, s, d); (void)s;
  float a = __expf(alpha_e[t] - alpha_max[d * HEADS + hh]);
  alpha_e[t] = a;
  atomicAdd(&alpha_sum[d * HEADS + hh], a);
}

// per (edge2, channel): weighted message, head-sum folded in, scattered to o_pre
__global__ void msg_kernel(const int* __restrict__ ei,
                           const float* __restrict__ xlr, const float* __restrict__ alpha_e,
                           const float* __restrict__ alpha_sum, float* __restrict__ o_pre) {
  long long t = (long long)blockIdx.x * blockDim.x + threadIdx.x;
  if (t >= (long long)NE2 * XD) return;
  int e2 = (int)(t / XD), c = (int)(t % XD);
  int s, d; edge2_sd(ei, e2, s, d);
  float a0 = alpha_e[e2 * 2 + 0] / alpha_sum[d * 2 + 0];
  float a1 = alpha_e[e2 * 2 + 1] / alpha_sum[d * 2 + 1];
  float m = a0 * xlr[(size_t)s * NCOL + c] + a1 * xlr[(size_t)s * NCOL + XD + c];
  atomicAdd(&o_pre[(size_t)d * XD + c], m);
}

// per (node, out j): head/degree mean + b_att, then @ W_fc1 + b_fc1
__global__ void final_kernel(const float* __restrict__ o_pre, const float* __restrict__ cnt,
                             const float* __restrict__ b_att, const float* __restrict__ Wfc1,
                             const float* __restrict__ bfc1, float* __restrict__ out) {
  int t = blockIdx.x * blockDim.x + threadIdx.x;
  if (t >= NN * 5) return;
  int n = t / 5, j = t % 5;
  float inv = 0.5f / (cnt[n] + 1.0f);        // mean over heads (2) and incoming edges (cnt+1 self loop)
  const float* __restrict__ op = o_pre + (size_t)n * XD;
  float acc = bfc1[j];
  for (int c = 0; c < XD; ++c)
    acc += (op[c] * inv + b_att[c]) * Wfc1[c * 5 + j];
  out[t] = acc;
}

// ---------------- launch ----------------
extern "C" void kernel_launch(void* const* d_in, const int* in_sizes, int n_in,
                              void* d_out, int out_size, void* d_ws, size_t ws_size,
                              hipStream_t stream) {
  const float* h        = (const float*)d_in[0];
  const float* edge_num = (const float*)d_in[1];
  const int*   ei       = (const int*)  d_in[2];
  const float* ew       = (const float*)d_in[3];
  const float* W_sum    = (const float*)d_in[4];
  const float* b_sum    = (const float*)d_in[5];
  const float* W_mean   = (const float*)d_in[6];
  const float* b_mean   = (const float*)d_in[7];
  const float* W_max    = (const float*)d_in[8];
  const float* b_max    = (const float*)d_in[9];
  const float* W_ne     = (const float*)d_in[10];
  const float* b_ne     = (const float*)d_in[11];
  const float* W_nem    = (const float*)d_in[12];
  const float* b_nem    = (const float*)d_in[13];
  const float* Wl       = (const float*)d_in[14];
  const float* Wr       = (const float*)d_in[15];
  const float* We       = (const float*)d_in[16];
  const float* att      = (const float*)d_in[17];
  const float* b_att    = (const float*)d_in[18];
  const float* W_fc1    = (const float*)d_in[19];
  const float* b_fc1    = (const float*)d_in[20];
  float* out = (float*)d_out;

  float* F = (float*)d_ws;
  float* acc_sum   = F + OFF_ACC_SUM;
  float* acc_mean  = F + OFF_ACC_MEAN;
  float* acc_ne    = F + OFF_ACC_NE;
  float* cnt       = F + OFF_CNT;
  float* ewsum     = F + OFF_EWSUM;
  float* alpha_sum = F + OFF_ALPHASUM;
  float* o_pre     = F + OFF_OPRE;
  float* acc_max   = F + OFF_ACC_MAX;
  float* acc_nem   = F + OFF_ACC_NEM;
  float* alpha_max = F + OFF_ALPHAMAX;
  float* alpha_e   = F + OFF_ALPHAE;
  float* xlr       = F + OFF_XLR;
  half_t* xpad = (half_t*)(F + OFF_HALF);
  half_t* Wt   = xpad + (size_t)NN * KPAD;

  const float ninf = -__builtin_huge_valf();

  // init: one big zero block, one -inf block
  fill_kernel<<<4096, 256, 0, stream>>>(F, 26000016ull, 0.0f);
  fill_kernel<<<2048, 256, 0, stream>>>(F + OFF_ACC_MAX, 6500000ull, ninf);

  // mean(edge_weight)
  ew_sum_kernel<<<(NE + 255) / 256, 256, 0, stream>>>(ew, ewsum);

  // GCN message scatter
  {
    long long nt = (long long)NE * CH;
    gcn_edge_kernel<<<(unsigned)((nt + 255) / 256), 256, 0, stream>>>(
        h, ei, ew, W_sum, W_mean, W_max, W_ne, W_nem,
        acc_sum, acc_mean, acc_max, acc_ne, acc_nem, cnt);
  }

  // build x (f16, K-padded) and transposed fused weights
  {
    long long nt = (long long)NN * KPAD;
    node_x_kernel<<<(unsigned)((nt + 255) / 256), 256, 0, stream>>>(
        edge_num, acc_sum, acc_mean, acc_max, acc_ne, acc_nem, cnt,
        b_sum, b_mean, b_max, b_ne, b_nem, xpad);
  }
  {
    long long nt = (long long)NCOL * KPAD;
    wt_kernel<<<(unsigned)((nt + 255) / 256), 256, 0, stream>>>(Wl, Wr, Wt);
  }

  // WMMA GEMM: xlr = x @ (Wl|Wr)
  {
    dim3 grid((MTILES + 7) / 8, (NCOL + 63) / 64);
    gemm_wmma_kernel<<<grid, 256, 0, stream>>>(xpad, Wt, xlr);
  }

  // GATv2 attention softmax + aggregation
  alpha1_kernel<<<(NE2 * HEADS + 255) / 256, 256, 0, stream>>>(ei, ew, ewsum, xlr, We, att,
                                                               alpha_e, alpha_max);
  alpha2_kernel<<<(NE2 * HEADS + 255) / 256, 256, 0, stream>>>(ei, alpha_e, alpha_max, alpha_sum);
  {
    long long nt = (long long)NE2 * XD;
    msg_kernel<<<(unsigned)((nt + 255) / 256), 256, 0, stream>>>(ei, xlr, alpha_e, alpha_sum, o_pre);
  }

  // final projection
  final_kernel<<<(NN * 5 + 255) / 256, 256, 0, stream>>>(o_pre, cnt, b_att, W_fc1, b_fc1, out);
}
